// LMHSA_39779987095846
// MI455X (gfx1250) — compile-verified, matrix operand order checked by reading, not run on hardware
//
#include <hip/hip_runtime.h>
#include <math.h>

typedef __bf16 bf16_t;
typedef __attribute__((ext_vector_type(16))) __bf16        v16bf;
typedef __attribute__((ext_vector_type(8)))  __bf16        v8bf;
typedef __attribute__((ext_vector_type(8)))  float         v8f;
typedef __attribute__((ext_vector_type(4)))  float         v4f;
typedef __attribute__((ext_vector_type(4)))  unsigned int  v4u;

#define HEADS 8
#define DK    32
#define DV    32
#define CCH   256
#define HH    56
#define WW    56
#define LQ    3136
#define LKV   784
#define KH    28
#define KW    28
#define BATCH 8
#define SCALE 0.17677669529663689f   // 32^-0.5

// ---------------- WMMA fragment helpers (gfx1250 bf16 layouts) ----------------
// 16-bit operand layout: lane r = lane&15, kbase = (lane>>4)*8;
// element e=0..7 -> K = kbase+e, e=8..15 -> K = kbase+16+(e-8).
// Works for A (rows in memory) and B^T-row-major (i.e. [N][K]) identically;
// both 8-element runs are contiguous -> two 16-byte loads.
struct Frag {
  union { v4u u[2]; v16bf v; };
};
union Pack8 { v4u u; v8bf v; };

__device__ __forceinline__ Frag load_frag(const bf16_t* base, int row0, int stride, int k0) {
  int lane = threadIdx.x & 31;
  int m    = lane & 15;
  int kb   = (lane >> 4) << 3;
  const bf16_t* p = base + (size_t)(row0 + m) * (size_t)stride + (size_t)(k0 + kb);
  Frag f;
  f.u[0] = *(const v4u*)(p);
  f.u[1] = *(const v4u*)(p + 16);
  return f;
}

__device__ __forceinline__ v8f wmma_bf16(const Frag& a, const Frag& b, v8f c) {
  return __builtin_amdgcn_wmma_f32_16x16x32_bf16(false, a.v, false, b.v, (short)0, c, false, false);
}

// ---------------- global mean/var (deterministic two-stage) ----------------
__global__ void reduce_stage1(const float* __restrict__ x, float2* __restrict__ part, size_t n) {
  __shared__ float ss[256], sq[256];
  float s = 0.f, q = 0.f;
  for (size_t i = (size_t)blockIdx.x * blockDim.x + threadIdx.x; i < n;
       i += (size_t)gridDim.x * blockDim.x) {
    float v = x[i]; s += v; q += v * v;
  }
  ss[threadIdx.x] = s; sq[threadIdx.x] = q;
  __syncthreads();
  for (int o = 128; o > 0; o >>= 1) {
    if ((int)threadIdx.x < o) {
      ss[threadIdx.x] += ss[threadIdx.x + o];
      sq[threadIdx.x] += sq[threadIdx.x + o];
    }
    __syncthreads();
  }
  if (threadIdx.x == 0) part[blockIdx.x] = make_float2(ss[0], sq[0]);
}

__global__ void reduce_stage2(const float2* __restrict__ part, float* __restrict__ stats, float invn) {
  __shared__ float ss[256], sq[256];
  float2 a = part[threadIdx.x];
  float2 b = part[threadIdx.x + 256];
  ss[threadIdx.x] = a.x + b.x; sq[threadIdx.x] = a.y + b.y;
  __syncthreads();
  for (int o = 128; o > 0; o >>= 1) {
    if ((int)threadIdx.x < o) {
      ss[threadIdx.x] += ss[threadIdx.x + o];
      sq[threadIdx.x] += sq[threadIdx.x + o];
    }
    __syncthreads();
  }
  if (threadIdx.x == 0) {
    float mu  = ss[0] * invn;
    float var = sq[0] * invn - mu * mu;
    stats[0] = mu;
    stats[1] = rsqrtf(var + 1e-5f);
  }
}

// ---------------- small converters ----------------
__global__ void f32_to_bf16_k(const float* __restrict__ in, bf16_t* __restrict__ o, int n) {
  int i = blockIdx.x * blockDim.x + threadIdx.x;
  if (i < n) o[i] = (bf16_t)in[i];
}

// xn[b][p][c] = (x[b][c][p]-mu)*rstd, bf16, GEMM-A layout
__global__ void xnorm_bf16(const float* __restrict__ x, const float* __restrict__ stats,
                           bf16_t* __restrict__ XN) {
  size_t idx = (size_t)blockIdx.x * blockDim.x + threadIdx.x;  // c fastest
  int c = (int)(idx & 255);
  size_t t = idx >> 8;
  int p = (int)(t % LQ);
  int b = (int)(t / LQ);
  float mu = stats[0], rs = stats[1];
  float v = (x[((size_t)b * CCH + c) * (HH * WW) + p] - mu) * rs;
  XN[idx] = (bf16_t)v;
}

// depthwise 3x3 stride-2 pad-1; output straight into GEMM-A layout [b][784][256] bf16
__global__ void dwconv_bf16(const float* __restrict__ x, const float* __restrict__ w,
                            const float* __restrict__ bias, bf16_t* __restrict__ outA) {
  int idx = blockIdx.x * blockDim.x + threadIdx.x;   // B*28*28*256
  int c = idx & 255;
  int t = idx >> 8;
  int ox = t % KW; t /= KW;
  int oy = t % KH;
  int b  = t / KH;
  const float* xp = x + ((size_t)b * CCH + c) * (HH * WW);
  const float* wp = w + c * 9;
  float s = bias[c];
#pragma unroll
  for (int dy = 0; dy < 3; ++dy) {
    int iy = oy * 2 - 1 + dy;
    if (iy < 0 || iy >= HH) continue;
#pragma unroll
    for (int dx = 0; dx < 3; ++dx) {
      int ix = ox * 2 - 1 + dx;
      if (ix < 0 || ix >= WW) continue;
      s += xp[iy * WW + ix] * wp[dy * 3 + dx];
    }
  }
  outA[((size_t)b * LKV + oy * KW + ox) * CCH + c] = (bf16_t)s;
}

// ---------------- bf16 WMMA GEMM: C[i][n] = A[i][:] . Bt[n][:] + bias[n] ------
// MODE 0: store bf16 row-major [b][Mdim][256]          (Q, K projections)
// MODE 1: store bf16 transposed Vt[b][h][d][784]       (V projection)
// MODE 2: fp32 out + flat residual x (fused final proj, out = gemm+bo+x)
template <int MODE>
__global__ void __launch_bounds__(256)
gemm16(const bf16_t* __restrict__ A, const bf16_t* __restrict__ Bt,
       const float* __restrict__ bias, void* __restrict__ outp,
       const float* __restrict__ xres, int Mdim) {
  int wv = blockIdx.x * (blockDim.x >> 5) + (threadIdx.x >> 5);
  int tilesM = Mdim >> 4;
  int wavesPerBatch = tilesM * (CCH / 16);
  int b   = wv / wavesPerBatch;
  int rem = wv % wavesPerBatch;
  int mt  = rem >> 4;
  int nt  = rem & 15;
  int i0 = mt << 4, n0 = nt << 4;

  const bf16_t* Ab = A + (size_t)b * Mdim * CCH;
  v8f acc = {};
#pragma unroll
  for (int kk = 0; kk < CCH / 32; ++kk) {
    Frag af = load_frag(Ab, i0, CCH, kk * 32);
    Frag bf = load_frag(Bt, n0, CCH, kk * 32);
    acc = wmma_bf16(af, bf, acc);
  }

  int lane = threadIdx.x & 31;
  int ncol = lane & 15;
  int rh   = (lane >> 4) << 3;
  float bvn = bias[n0 + ncol];
#pragma unroll
  for (int r = 0; r < 8; ++r) {
    int row = i0 + r + rh;
    int col = n0 + ncol;
    float val = acc[r] + bvn;
    if (MODE == 0) {
      ((bf16_t*)outp)[((size_t)b * Mdim + row) * CCH + col] = (bf16_t)val;
    } else if (MODE == 1) {
      int h = col >> 5, d = col & 31;
      ((bf16_t*)outp)[(((size_t)b * HEADS + h) * DV + d) * LKV + row] = (bf16_t)val;
    } else {
      size_t idx = (size_t)b * ((size_t)LQ * CCH) + (size_t)row * CCH + col;
      ((float*)outp)[idx] = val + xres[idx];
    }
  }
}

// ---------------- flash attention, transposed-score formulation ----------------
// Per wave: 16 query rows of one (b,h). Compute S^T = K x Q^T so that
//   - j lives across VGPR slots (+ half-wave), i lives across lanes
//   - softmax stats are scalar-per-lane, reductions are in-lane + ONE shfl_xor(16)
//   - P^T in registers is exactly the B-fragment layout for O^T = V^T x P^T
//     (element e<8 <- St1 slot e, e>=8 <- St2 slot e): no LDS transpose at all.
__global__ void __launch_bounds__(256)
attn_fa(const bf16_t* __restrict__ Q, const bf16_t* __restrict__ K,
        const bf16_t* __restrict__ Vt, const float* __restrict__ Bbias,
        bf16_t* __restrict__ RES) {
  int wv  = threadIdx.x >> 5;
  int gw  = blockIdx.x * 8 + wv;
  int b   = gw / (HEADS * (LQ / 16));
  int rem = gw % (HEADS * (LQ / 16));
  int h   = rem / (LQ / 16);
  int it  = rem % (LQ / 16);
  int i0  = it << 4;
  int lane = threadIdx.x & 31;
  int ncol = lane & 15;           // query column i - i0 owned by this lane
  int joff = (lane >> 4) << 3;    // 0 (lanes 0-15) or 8 (lanes 16-31)

  const bf16_t* Qb = Q  + ((size_t)b * LQ) * CCH + h * DK;
  const bf16_t* Kb = K  + ((size_t)b * LKV) * CCH + h * DK;
  const bf16_t* Vb = Vt + (((size_t)b * HEADS + h) * DV) * LKV;

  Frag qf = load_frag(Qb, i0, CCH, 0);          // B operand (Q^T), loaded once

  // per-lane contiguous bias run: Bbias[h][i0+ncol][j0+joff .. +7]
  const float* bbase = Bbias + ((size_t)h * LQ + i0 + ncol) * LKV + joff;

  float m = -3.0e38f, l = 0.f;
  v8f o1 = {}, o2 = {};                          // O^T rows d=0..15 / 16..31
  v8f zero = {};

  for (int jt = 0; jt < (LKV + 31) / 32; ++jt) {
    int j0 = jt * 32;
    bool full = (j0 + 32) <= LKV;                // uniform; tail = last 16 keys

    Frag k1 = load_frag(Kb, j0, CCH, 0);         // A: K rows j0..j0+15
    v8f s1 = wmma_bf16(k1, qf, zero);            // S^T[j][i]
    v8f s2 = zero;
    if (full) {
      Frag k2 = load_frag(Kb, j0 + 16, CCH, 0);
      s2 = wmma_bf16(k2, qf, zero);
    }

    float sv1[8], sv2[8];
    v4f ba0 = *(const v4f*)(bbase + j0);
    v4f ba1 = *(const v4f*)(bbase + j0 + 4);
#pragma unroll
    for (int r = 0; r < 8; ++r)
      sv1[r] = s1[r] * SCALE + (r < 4 ? ba0[r] : ba1[r - 4]);
    if (full) {
      v4f bb0 = *(const v4f*)(bbase + j0 + 16);
      v4f bb1 = *(const v4f*)(bbase + j0 + 20);
#pragma unroll
      for (int r = 0; r < 8; ++r)
        sv2[r] = s2[r] * SCALE + (r < 4 ? bb0[r] : bb1[r - 4]);
    }

    // row max over j: in-lane over 8 (or 16) slots, then merge half-waves
    float mx = sv1[0];
#pragma unroll
    for (int r = 1; r < 8; ++r) mx = fmaxf(mx, sv1[r]);
    if (full) {
#pragma unroll
      for (int r = 0; r < 8; ++r) mx = fmaxf(mx, sv2[r]);
    }
    mx = fmaxf(mx, __shfl_xor(mx, 16, 32));
    float mn = fmaxf(m, mx);
    float al = __expf(m - mn);
    m = mn;

    // P^T: exp in place; registers already match the PV B-fragment layout
    Frag pf;
    float sm = 0.f;
#pragma unroll
    for (int r = 0; r < 8; ++r) {
      float p = __expf(sv1[r] - mn);
      sm += p;
      pf.v[r] = (bf16_t)p;
    }
#pragma unroll
    for (int r = 0; r < 8; ++r) {
      float p = full ? __expf(sv2[r] - mn) : 0.f;
      sm += p;
      pf.v[8 + r] = (bf16_t)p;
    }
    sm += __shfl_xor(sm, 16, 32);
    l = l * al + sm;

#pragma unroll
    for (int r = 0; r < 8; ++r) { o1[r] *= al; o2[r] *= al; }

    Frag v1 = load_frag(Vb, 0,  LKV, j0);        // A: Vt rows d=0..15, K=j
    Frag v2 = load_frag(Vb, 16, LKV, j0);        // tail K>784 hits zeroed pad
    o1 = wmma_bf16(v1, pf, o1);
    o2 = wmma_bf16(v2, pf, o2);
  }

  // O^T[d][i]: per lane i fixed, 8 contiguous d per tile -> two packed stores
  float inv = 1.0f / l;
  bf16_t* ob = RES + ((size_t)b * LQ + i0 + ncol) * CCH + h * DV + joff;
  Pack8 pk1, pk2;
#pragma unroll
  for (int r = 0; r < 8; ++r) {
    pk1.v[r] = (bf16_t)(o1[r] * inv);
    pk2.v[r] = (bf16_t)(o2[r] * inv);
  }
  *(v4u*)(ob)      = pk1.u;
  *(v4u*)(ob + 16) = pk2.u;
}

// ---------------- launcher ----------------
extern "C" void kernel_launch(void* const* d_in, const int* in_sizes, int n_in,
                              void* d_out, int out_size, void* d_ws, size_t ws_size,
                              hipStream_t stream) {
  (void)in_sizes; (void)n_in; (void)out_size; (void)ws_size;
  const float* x     = (const float*)d_in[0];
  const float* Wk_dw = (const float*)d_in[1];
  const float* bk_dw = (const float*)d_in[2];
  const float* Wv_dw = (const float*)d_in[3];
  const float* bv_dw = (const float*)d_in[4];
  const float* Wq    = (const float*)d_in[5];
  const float* bq    = (const float*)d_in[6];
  const float* Wk    = (const float*)d_in[7];
  const float* bk    = (const float*)d_in[8];
  const float* Wv    = (const float*)d_in[9];
  const float* bv    = (const float*)d_in[10];
  const float* Wo    = (const float*)d_in[11];
  const float* bo    = (const float*)d_in[12];
  const float* Bbias = (const float*)d_in[13];
  float* out = (float*)d_out;

  char* ws = (char*)d_ws;
  size_t off = 0;
  auto take = [&](size_t bytes) -> void* {
    off = (off + 255) & ~(size_t)255;
    void* p = (void*)(ws + off);
    off += bytes;
    return p;
  };
  const size_t XN_BYTES = (size_t)BATCH * LQ  * CCH * 2;   // 12.85 MB
  const size_t KV_BYTES = (size_t)BATCH * LKV * CCH * 2;   //  3.21 MB
  const size_t PAD      = 8192;

  float2* part  = (float2*)take(512 * sizeof(float2));
  float*  stats = (float*) take(16 * sizeof(float));
  bf16_t* WqB   = (bf16_t*)take(CCH * CCH * 2);
  bf16_t* WkB   = (bf16_t*)take(CCH * CCH * 2);
  bf16_t* WvB   = (bf16_t*)take(CCH * CCH * 2);
  bf16_t* WoB   = (bf16_t*)take(CCH * CCH * 2);
  bf16_t* XN    = (bf16_t*)take(XN_BYTES);
  bf16_t* KF    = (bf16_t*)take(KV_BYTES);
  bf16_t* VF    = (bf16_t*)take(KV_BYTES);
  bf16_t* Qb    = (bf16_t*)take(XN_BYTES);
  bf16_t* Kb    = (bf16_t*)take(KV_BYTES + PAD);
  bf16_t* Vtb   = (bf16_t*)take(KV_BYTES + PAD);
  bf16_t* RES   = (bf16_t*)take(XN_BYTES);

  // zero tail pads: keeps masked-tile V fragments finite (0 * finite == 0)
  hipMemsetAsync((char*)Kb  + KV_BYTES, 0, PAD, stream);
  hipMemsetAsync((char*)Vtb + KV_BYTES, 0, PAD, stream);

  const size_t nelem = (size_t)BATCH * CCH * HH * WW;      // 6,422,528
  reduce_stage1<<<512, 256, 0, stream>>>(x, part, nelem);
  reduce_stage2<<<1, 256, 0, stream>>>(part, stats, 1.0f / (float)nelem);

  f32_to_bf16_k<<<256, 256, 0, stream>>>(Wq, WqB, CCH * CCH);
  f32_to_bf16_k<<<256, 256, 0, stream>>>(Wk, WkB, CCH * CCH);
  f32_to_bf16_k<<<256, 256, 0, stream>>>(Wv, WvB, CCH * CCH);
  f32_to_bf16_k<<<256, 256, 0, stream>>>(Wo, WoB, CCH * CCH);

  xnorm_bf16<<<(int)(nelem / 256), 256, 0, stream>>>(x, stats, XN);
  dwconv_bf16<<<(BATCH * KH * KW * CCH) / 256, 256, 0, stream>>>(x, Wk_dw, bk_dw, KF);
  dwconv_bf16<<<(BATCH * KH * KW * CCH) / 256, 256, 0, stream>>>(x, Wv_dw, bv_dw, VF);

  gemm16<0><<<(BATCH * (LQ  / 16) * 16) / 8, 256, 0, stream>>>(XN, WqB, bq, (void*)Qb,  nullptr, LQ);
  gemm16<0><<<(BATCH * (LKV / 16) * 16) / 8, 256, 0, stream>>>(KF, WkB, bk, (void*)Kb,  nullptr, LKV);
  gemm16<1><<<(BATCH * (LKV / 16) * 16) / 8, 256, 0, stream>>>(VF, WvB, bv, (void*)Vtb, nullptr, LKV);

  attn_fa<<<(BATCH * HEADS * (LQ / 16)) / 8, 256, 0, stream>>>(Qb, Kb, Vtb, Bbias, RES);

  gemm16<2><<<(BATCH * (LQ / 16) * 16) / 8, 256, 0, stream>>>(RES, WoB, bo, (void*)out, x, LQ);
}
